// RepLKBlock_57655640981953
// MI455X (gfx1250) — matrix operand order repaired and checked by reading, not compiled, not run on hardware
//
#include <hip/hip_runtime.h>
#include <math.h>

#define NB    16
#define CDIM  192
#define Hh    56
#define Ww    56
#define HW    (Hh*Ww)          // 3136
#define KL    21
#define KS    5
#define PAD   10
#define HP    (Hh + 2*PAD)     // 76
#define GSTRH 200              // LDS row stride (bf16 elems) for G tiles: 400B, 16B-aligned,
                               // 100 dwords -> lane bank offsets p*36 mod 64 all distinct

typedef __attribute__((ext_vector_type(16))) __bf16 v16bf;
typedef __attribute__((ext_vector_type(8)))  float  v8f;

__device__ __forceinline__ unsigned short f2bf_rne(float f) {
  unsigned int u = __builtin_bit_cast(unsigned int, f);
  unsigned int r = u + 0x7FFFu + ((u >> 16) & 1u);
  return (unsigned short)(r >> 16);
}
__device__ __forceinline__ float bf2f(unsigned short h) {
  unsigned int u = ((unsigned int)h) << 16;
  return __builtin_bit_cast(float, u);
}

union FragBF { uint4 u[2]; v16bf v; };

// ---------------------------------------------------------------------------
// Kernel 0: split w_pw (192x192 f32, [co][ci]) into hi/lo bf16, row-major.
// ---------------------------------------------------------------------------
__global__ __launch_bounds__(256) void repack_w_kernel(
    const float* __restrict__ wpw, unsigned short* __restrict__ wh,
    unsigned short* __restrict__ wl2)
{
  const int i = blockIdx.x * 256 + threadIdx.x;
  if (i < CDIM * CDIM) {
    const float v = wpw[i];
    const unsigned short h = f2bf_rne(v);
    wh[i]  = h;
    wl2[i] = f2bf_rne(v - bf2f(h));
  }
}

// ---------------------------------------------------------------------------
// Kernel 1: fused depthwise 21x21 + 5x5 (folded into center), per-(n,c) plane.
// Two adjacent pixels/thread with a rolling register window: 22 LDS reads feed
// 42 FMAs per tap-row, two independent chains for VOPD dual-issue.
// ---------------------------------------------------------------------------
__global__ __launch_bounds__(256) void dwconv_kernel(
    const float* __restrict__ x, const float* __restrict__ wl,
    const float* __restrict__ ws, float* __restrict__ out1,
    float* __restrict__ partial)
{
  __shared__ float sx[HP * HP];     // 76x76 halo plane, 23.1 KB
  __shared__ float sw[KL * KL];     // combined 21x21 weights
  __shared__ float red[2][8];

  const int c = blockIdx.x, n = blockIdx.y;
  const int tid = threadIdx.x;
  const float* __restrict__ xp = x + ((size_t)(n * CDIM + c)) * HW;

  for (int i = tid; i < HP * HP; i += 256) {
    int hh = i / HP - PAD, ww = i % HP - PAD;
    float v = 0.0f;
    if (hh >= 0 && hh < Hh && ww >= 0 && ww < Ww) v = xp[hh * Ww + ww];
    sx[i] = v;
  }
  for (int i = tid; i < KL * KL; i += 256) {
    int kh = i / KL, kw = i % KL;
    float v = wl[c * KL * KL + i];
    int dh = kh - (KL / 2 - KS / 2), dw = kw - (KL / 2 - KS / 2);
    if (dh >= 0 && dh < KS && dw >= 0 && dw < KS)
      v += ws[c * KS * KS + dh * KS + dw];
    sw[i] = v;
  }
  __syncthreads();

  float s1 = 0.0f, s2 = 0.0f;
  for (int pp = tid; pp < HW / 2; pp += 256) {
    const int h  = pp / (Ww / 2);
    const int w2 = (pp % (Ww / 2)) * 2;
    const float* __restrict__ base = sx + h * HP + w2;
    float a0 = 0.0f, a1 = 0.0f;
    for (int kh = 0; kh < KL; ++kh) {
      const float* __restrict__ r  = base + kh * HP;
      const float* __restrict__ wr = sw + kh * KL;   // uniform broadcast reads
      float prev = r[0];
#pragma unroll
      for (int kw = 0; kw < KL; ++kw) {
        const float cur = r[kw + 1];
        const float wv  = wr[kw];
        a0 = fmaf(wv, prev, a0);
        a1 = fmaf(wv, cur, a1);
        prev = cur;
      }
    }
    const size_t o = ((size_t)(n * CDIM + c)) * HW + h * Ww + w2;
    out1[o]     = a0;
    out1[o + 1] = a1;
    s1 += a0 + a1;
    s2 += fmaf(a0, a0, a1 * a1);
  }

  for (int off = 16; off > 0; off >>= 1) {
    s1 += __shfl_xor(s1, off, 32);
    s2 += __shfl_xor(s2, off, 32);
  }
  const int wv = tid >> 5, ln = tid & 31;
  if (ln == 0) { red[0][wv] = s1; red[1][wv] = s2; }
  __syncthreads();
  if (tid == 0) {
    float a = 0.0f, b = 0.0f;
    for (int i = 0; i < 8; ++i) { a += red[0][i]; b += red[1][i]; }
    partial[(n * CDIM + c) * 2 + 0] = a;
    partial[(n * CDIM + c) * 2 + 1] = b;
  }
}

// ---------------------------------------------------------------------------
// Kernel 2: per-sample mean/rstd, one wave per sample, fixed order.
// ---------------------------------------------------------------------------
__global__ __launch_bounds__(32) void stats_kernel(
    const float* __restrict__ partial, float* __restrict__ stats)
{
  const int n = blockIdx.x, ln = threadIdx.x;
  float s1 = 0.0f, s2 = 0.0f;
  for (int i = ln; i < CDIM; i += 32) {
    s1 += partial[(n * CDIM + i) * 2 + 0];
    s2 += partial[(n * CDIM + i) * 2 + 1];
  }
  for (int off = 16; off > 0; off >>= 1) {
    s1 += __shfl_xor(s1, off, 32);
    s2 += __shfl_xor(s2, off, 32);
  }
  if (ln == 0) {
    const float inv = 1.0f / (float)(CDIM * HW);
    const float mean = s1 * inv;
    const float var = s2 * inv - mean * mean;
    stats[n * 2 + 0] = mean;
    stats[n * 2 + 1] = rsqrtf(var + 1e-5f);
  }
}

// ---------------------------------------------------------------------------
// Kernel 3: norm + exact GELU + 1x1 conv as split-bf16 WMMA GEMM + bias + res.
// Block = 12 waves (384 thr), one (n, 64-pixel) tile; wave w owns co-tile w.
// D = Wh*Gh + Wh*Gl + Wl*Gh, f32 accumulate: 3x v_wmma_f32_16x16x32_bf16 per
// K=32 chunk per 16x16 sub-tile; 6 K-chunks x 4 pixel sub-tiles per wave.
// ---------------------------------------------------------------------------
__global__ __launch_bounds__(384) void pw_gemm_kernel(
    const float* __restrict__ out1, const float* __restrict__ x,
    const float* __restrict__ gamma, const float* __restrict__ beta,
    const unsigned short* __restrict__ wh, const unsigned short* __restrict__ wl2,
    const float* __restrict__ bpw, const float* __restrict__ stats,
    float* __restrict__ out)
{
  // K-contiguous activation tiles: gT[pixel 0..63][ci 0..191], hi and lo parts
  __shared__ unsigned short gh[64 * GSTRH];   // 25,600 B
  __shared__ unsigned short gl[64 * GSTRH];   // 25,600 B

  const int pt = blockIdx.x;        // 0..48 pixel tiles of 64
  const int n  = blockIdx.y;
  const int tid = threadIdx.x;
  const int p0 = pt * 64;

  const float mean = stats[n * 2 + 0];
  const float rstd = stats[n * 2 + 1];
  const float* __restrict__ o1 = out1 + (size_t)n * CDIM * HW;

  // Stage normalized + GELU activations, split into hi/lo bf16.
  // Consecutive threads -> consecutive pixels j: coalesced global reads.
  for (int i = tid; i < CDIM * 64; i += 384) {
    const int ci = i >> 6, j = i & 63;
    const float v = o1[ci * HW + p0 + j];
    const float a = (v - mean) * rstd * gamma[ci] + beta[ci];
    const float ge = 0.5f * a * (1.0f + erff(a * 0.70710678118654752f));
    const unsigned short h = f2bf_rne(ge);
    gh[j * GSTRH + ci] = h;
    gl[j * GSTRH + ci] = f2bf_rne(ge - bf2f(h));
  }
  __syncthreads();

  const int wv = tid >> 5;          // 0..11 -> co tile
  const int ln = tid & 31;
  const int co0 = wv * 16;
  const int mrow = ln & 15;         // M (A) / N (B,C,D) index within half-wave
  const int khalf = ln >> 4;        // half-wave selector

  v8f acc[4] = {v8f{}, v8f{}, v8f{}, v8f{}};

  // A-fragment base: row co0+mrow of W (bf16, 384 B row stride). Contiguous
  // layout matching B: V0-7 <- K = 16h..16h+15 (32 B).
  const char* wh_row = (const char*)wh  + ((size_t)(co0 + mrow) * CDIM + 16 * khalf) * 2;
  const char* wl_row = (const char*)wl2 + ((size_t)(co0 + mrow) * CDIM + 16 * khalf) * 2;

  for (int k0 = 0; k0 < CDIM; k0 += 32) {
    FragBF ah, al;
    ah.u[0] = *(const uint4*)(wh_row + 2 * k0);
    ah.u[1] = *(const uint4*)(wh_row + 2 * k0 + 16);
    al.u[0] = *(const uint4*)(wl_row + 2 * k0);
    al.u[1] = *(const uint4*)(wl_row + 2 * k0 + 16);

#pragma unroll
    for (int ps = 0; ps < 4; ++ps) {
      // B fragment: contiguous K order, same as A = lanes 0-15 hold K 0..15, lanes
      // 16-31 hold K 16..31, packed 2/VGPR in increasing K -> one contiguous
      // 32-byte per-lane chunk of column (pixel) p.
      const int p = ps * 16 + mrow;
      const char* gh_p = (const char*)gh + ((size_t)p * GSTRH + k0 + 16 * khalf) * 2;
      const char* gl_p = (const char*)gl + ((size_t)p * GSTRH + k0 + 16 * khalf) * 2;
      FragBF bh, bl;
      bh.u[0] = *(const uint4*)(gh_p);
      bh.u[1] = *(const uint4*)(gh_p + 16);
      bl.u[0] = *(const uint4*)(gl_p);
      bl.u[1] = *(const uint4*)(gl_p + 16);

      acc[ps] = __builtin_amdgcn_wmma_f32_16x16x32_bf16(
          false, ah.v, false, bh.v, (short)0, acc[ps], false, false);
      acc[ps] = __builtin_amdgcn_wmma_f32_16x16x32_bf16(
          false, ah.v, false, bl.v, (short)0, acc[ps], false, false);
      acc[ps] = __builtin_amdgcn_wmma_f32_16x16x32_bf16(
          false, al.v, false, bh.v, (short)0, acc[ps], false, false);
    }
  }

  // Epilogue: + bias + residual, coalesced stores (16 consecutive px per row)
#pragma unroll
  for (int ps = 0; ps < 4; ++ps) {
    const int p = p0 + ps * 16 + mrow;
#pragma unroll
    for (int r = 0; r < 8; ++r) {
      const int co = co0 + khalf * 8 + r;
      const size_t idx = ((size_t)n * CDIM + co) * HW + p;
      out[idx] = acc[ps][r] + bpw[co] + x[idx];
    }
  }
}

// ---------------------------------------------------------------------------
extern "C" void kernel_launch(void* const* d_in, const int* in_sizes, int n_in,
                              void* d_out, int out_size, void* d_ws, size_t ws_size,
                              hipStream_t stream) {
  const float* x   = (const float*)d_in[0];
  const float* wl  = (const float*)d_in[1];
  const float* ws  = (const float*)d_in[2];
  const float* gam = (const float*)d_in[3];
  const float* bet = (const float*)d_in[4];
  const float* wpw = (const float*)d_in[5];
  const float* bpw = (const float*)d_in[6];
  float* out = (float*)d_out;

  float* out1    = (float*)d_ws;                           // N*C*H*W f32
  float* partial = out1 + (size_t)NB * CDIM * HW;          // N*C*2 f32
  float* stats   = partial + (size_t)NB * CDIM * 2;        // N*2 f32
  unsigned short* wh  = (unsigned short*)(stats + NB * 2); // 192*192 bf16 hi
  unsigned short* wl2 = wh + (size_t)CDIM * CDIM;          // 192*192 bf16 lo

  repack_w_kernel<<<(CDIM * CDIM + 255) / 256, 256, 0, stream>>>(wpw, wh, wl2);

  dim3 gridA(CDIM, NB);
  dwconv_kernel<<<gridA, 256, 0, stream>>>(x, wl, ws, out1, partial);

  stats_kernel<<<NB, 32, 0, stream>>>(partial, stats);

  dim3 gridC(HW / 64, NB);  // 49 x 16
  pw_gemm_kernel<<<gridC, 384, 0, stream>>>(out1, x, gam, bet, wh, wl2, bpw, stats, out);
}
